// Net_87814901334404
// MI455X (gfx1250) — matrix operand order repaired and checked by reading, not compiled, not run on hardware
//
#include <hip/hip_runtime.h>

// ---------------- model constants (match reference) ----------------
static constexpr int Bb = 512, CI = 3, HW = 28, KK = 7, CC = 1024, NHh = 16, LL = 12;
static constexpr int TT = 1 + (HW / KK) * (HW / KK);   // 17 tokens
static constexpr int MM = Bb * TT;                     // 8704 rows
static constexpr int HD = CC / NHh;                    // 64 head dim
static constexpr int PATCH = CI * KK * KK;             // 147

typedef __attribute__((ext_vector_type(16))) __bf16 v16bf;
typedef __attribute__((ext_vector_type(8)))  float  v8f;

__device__ __forceinline__ unsigned short f32_to_bf16(float f) {
  unsigned u = __float_as_uint(f);
  return (unsigned short)((u + 0x7FFFu + ((u >> 16) & 1u)) >> 16);
}

// CDNA5 async global->LDS copy (VGLOBAL GLOBAL_LOAD_ASYNC_TO_LDS_B128, ASYNCcnt).
// LDS operand = low 32 bits of generic shared pointer.
__device__ __forceinline__ void async_copy_b128(void* lds_dst, const void* gsrc) {
  unsigned l = (unsigned)(unsigned long long)lds_dst;
  asm volatile("global_load_async_to_lds_b128 %0, %1, off"
               :: "v"(l), "v"(gsrc) : "memory");
}
__device__ __forceinline__ void wait_asynccnt0() {
  asm volatile("s_wait_asynccnt 0x0" ::: "memory");
}

// ---------------- fp32 -> bf16 elementwise (weight pre-conversion) ----------------
__global__ __launch_bounds__(256) void cvt_bf16_kernel(
    const float* __restrict__ src, unsigned short* __restrict__ dst)
{
  int i = (blockIdx.x * 256 + threadIdx.x) * 4;
  float4 f = *(const float4*)(src + i);
  union { unsigned short us[4]; uint2 u2; } pk;
  pk.us[0] = f32_to_bf16(f.x); pk.us[1] = f32_to_bf16(f.y);
  pk.us[2] = f32_to_bf16(f.z); pk.us[3] = f32_to_bf16(f.w);
  *(uint2*)(dst + i) = pk.u2;
}

// ---------------- embed: thumbnail + patchify conv (kernel==stride) ----------------
__global__ __launch_bounds__(256) void embed_kernel(
    const float* __restrict__ x, const float* __restrict__ cw, float* __restrict__ h)
{
  __shared__ float patch[PATCH + 13];
  int bt = blockIdx.x;
  int t = bt % TT, b = bt / TT;
  int tid = threadIdx.x;
  if (tid < PATCH) {
    int c = tid / 49, r = tid % 49, p = r / 7, qq = r % 7;
    const float* xb = x + ((size_t)(b * CI + c) * HW) * HW;
    float val;
    if (t == 0) {
      // bilinear 28->7, half-pixel: out p maps to in 4p+1.5 -> mean of 2x2 at (4p+1,4q+1)
      int rr = 4 * p + 1, cc2 = 4 * qq + 1;
      val = 0.25f * (xb[rr * HW + cc2] + xb[rr * HW + cc2 + 1] +
                     xb[(rr + 1) * HW + cc2] + xb[(rr + 1) * HW + cc2 + 1]);
    } else {
      int ti = (t - 1) / 4, tj = (t - 1) % 4;
      val = xb[(ti * KK + p) * HW + tj * KK + qq];
    }
    patch[tid] = val;
  }
  __syncthreads();
#pragma unroll
  for (int dd = 0; dd < 4; ++dd) {
    int d = tid * 4 + dd;                    // 0..1023
    const float* wrow = cw + (size_t)d * PATCH;
    float acc = 0.f;
#pragma unroll 7
    for (int j = 0; j < PATCH; ++j) acc += patch[j] * wrow[j];
    h[(size_t)bt * CC + d] = acc;
  }
}

// ---------------- fused residual-add + LayerNorm -> bf16 ----------------
__global__ __launch_bounds__(256) void ln_res_kernel(
    float* __restrict__ h, const float* __restrict__ res,
    const float* __restrict__ w, const float* __restrict__ b,
    unsigned short* __restrict__ outb, int do_ln)
{
  int wave = threadIdx.x >> 5, lane = threadIdx.x & 31;
  int row = blockIdx.x * 8 + wave;
  float* hp = h + (size_t)row * CC;
  const float* rp = res ? res + (size_t)row * CC : nullptr;
  float vals[CC / 32];
  float s = 0.f, s2 = 0.f;
#pragma unroll
  for (int i = 0; i < CC / 32; ++i) {
    int c = lane + i * 32;
    float xv = hp[c];
    if (rp) { xv += rp[c]; hp[c] = xv; }
    vals[i] = xv; s += xv; s2 += xv * xv;
  }
#pragma unroll
  for (int off = 16; off; off >>= 1) {
    s  += __shfl_xor(s, off, 32);
    s2 += __shfl_xor(s2, off, 32);
  }
  float mean = s * (1.f / CC);
  float var  = s2 * (1.f / CC) - mean * mean;
  float rstd = rsqrtf(var + 1e-5f);
  unsigned short* op = outb + (size_t)row * CC;
#pragma unroll
  for (int i = 0; i < CC / 32; ++i) {
    int c = lane + i * 32;
    float y = do_ln ? ((vals[i] - mean) * rstd * w[c] + b[c]) : vals[i];
    op[c] = f32_to_bf16(y);
  }
}

// ---------------- WMMA GEMM: O[M,N] = A[M,K](bf16) @ Wb[N,K]^T(bf16) + bias ----------------
// 256 thr (8 waves), workgroup tile 128x128, wave tile 32x64 (8 WMMAs), K step 32,
// double-buffered LDS with ASYNC global->LDS staging (no VGPR staging, no spills),
// one barrier per K-step, v_wmma_f32_16x16x32_bf16.
union Frag16 { uint4 u[2]; v16bf v; };

__global__ __launch_bounds__(256) void gemm_bf16(
    const unsigned short* __restrict__ A, const unsigned short* __restrict__ Wb,
    const float* __restrict__ bias, float* __restrict__ O, int N, int K)
{
  __shared__ unsigned short lA[2][128 * 32];   // ping-pong A tiles (2 x 8KB)
  __shared__ unsigned short lB[2][128 * 32];   // ping-pong B tiles (2 x 8KB)

  const int tid  = threadIdx.x;
  const int wave = tid >> 5, lane = tid & 31;
  const int m0 = blockIdx.x * 128, n0 = blockIdx.y * 128;
  const int wm = (wave >> 1) * 32;          // 0,32,64,96
  const int wn = (wave & 1) * 64;           // 0,64
  const int half = lane >> 4, lrow = lane & 15;

  // staging decomposition: 128 rows x 4 chunks of 16B (8 bf16) each
  const int row0 = tid >> 2,           c0 = tid & 3;
  const int row1 = (tid + 256) >> 2,   c1 = (tid + 256) & 3;

  v8f acc[2][4] = {};

  auto stage_async = [&](int buf, int k0) {
    async_copy_b128(lA[buf] + row0 * 32 + c0 * 8, A  + (size_t)(m0 + row0) * K + k0 + c0 * 8);
    async_copy_b128(lA[buf] + row1 * 32 + c1 * 8, A  + (size_t)(m0 + row1) * K + k0 + c1 * 8);
    async_copy_b128(lB[buf] + row0 * 32 + c0 * 8, Wb + (size_t)(n0 + row0) * K + k0 + c0 * 8);
    async_copy_b128(lB[buf] + row1 * 32 + c1 * 8, Wb + (size_t)(n0 + row1) * K + k0 + c1 * 8);
  };

  auto compute = [&](int buf) {
    Frag16 fa2[2], fb2[4];
    // A fragment (16-bit A 16x32): lane<16 -> elems {0..7,16..23}; lane>=16 -> {8..15,24..31}
#pragma unroll
    for (int mi = 0; mi < 2; ++mi) {
      const unsigned short* rp = lA[buf] + (wm + mi * 16 + lrow) * 32;
      fa2[mi].u[0] = *(const uint4*)(rp + (half ? 8 : 0));
      fa2[mi].u[1] = *(const uint4*)(rp + (half ? 24 : 16));
    }
    // B fragment (32x16, lane = N col): lane<16 -> K {0..7,16..23}; lane>=16 -> K {8..15,24..31}
#pragma unroll
    for (int ni = 0; ni < 4; ++ni) {
      const unsigned short* rp = lB[buf] + (wn + ni * 16 + lrow) * 32;
      fb2[ni].u[0] = *(const uint4*)(rp + (half ? 8 : 0));
      fb2[ni].u[1] = *(const uint4*)(rp + (half ? 24 : 16));
    }
#pragma unroll
    for (int mi = 0; mi < 2; ++mi)
#pragma unroll
      for (int ni = 0; ni < 4; ++ni)
        acc[mi][ni] = __builtin_amdgcn_wmma_f32_16x16x32_bf16(
            false, fa2[mi].v, false, fb2[ni].v, (short)0, acc[mi][ni], false, false);
  };

  // ping-pong pipeline: async copies into inactive buffer overlap WMMAs on active one
  stage_async(0, 0);
  wait_asynccnt0();
  __syncthreads();
  int cur = 0;
  for (int k0 = 32; k0 < K; k0 += 32) {
    stage_async(cur ^ 1, k0);  // memory -> LDS, no VGPR staging
    compute(cur);              // 8 WMMAs + fragment ds_loads
    wait_asynccnt0();          // own copies committed to LDS
    __syncthreads();           // everyone's copies visible
    cur ^= 1;
  }
  compute(cur);

  // C/D layout: VGPR j -> M = j + 8*(lane>=16), N = lane%16
#pragma unroll
  for (int mi = 0; mi < 2; ++mi)
#pragma unroll
    for (int ni = 0; ni < 4; ++ni) {
      int colg = n0 + wn + ni * 16 + lrow;
      float bv = bias ? bias[colg] : 0.f;
#pragma unroll
      for (int j = 0; j < 8; ++j) {
        int rowg = m0 + wm + mi * 16 + j + half * 8;
        O[(size_t)rowg * N + colg] = acc[mi][ni][j] + bv;
      }
    }
}

// ---------------- attention: one wave per (b, head, query row); T=17, hd=64 ----------------
__global__ __launch_bounds__(256) void attn_kernel(
    const float* __restrict__ q, const float* __restrict__ k,
    const float* __restrict__ v, float* __restrict__ o)
{
  int wave = threadIdx.x >> 5, lane = threadIdx.x & 31;
  int gw = blockIdx.x * 8 + wave;           // over B*NH*T
  int qi = gw % TT;
  int rem = gw / TT;
  int hh = rem % NHh;
  int bb = rem / NHh;
  const size_t base = (size_t)bb * TT * CC + (size_t)hh * HD;
  const float* qp = q + base + (size_t)qi * CC;
  float q0 = qp[lane], q1 = qp[lane + 32];
  const float scale = 0.125f;               // 1/sqrt(64)
  float sc[TT];
  float mx = -3.0e38f;
#pragma unroll
  for (int t2 = 0; t2 < TT; ++t2) {
    float p = -3.0e38f;
    if (t2 <= qi) {                         // causal; uniform across wave
      const float* kp = k + base + (size_t)t2 * CC;
      float d = q0 * kp[lane] + q1 * kp[lane + 32];
      d += __shfl_xor(d, 16, 32); d += __shfl_xor(d, 8, 32);
      d += __shfl_xor(d, 4, 32);  d += __shfl_xor(d, 2, 32);
      d += __shfl_xor(d, 1, 32);
      p = d * scale;
    }
    sc[t2] = p; mx = fmaxf(mx, p);
  }
  float se = 0.f;
#pragma unroll
  for (int t2 = 0; t2 < TT; ++t2) {
    float e = (t2 <= qi) ? __expf(sc[t2] - mx) : 0.f;
    sc[t2] = e; se += e;
  }
  float inv = 1.f / se;
  float o0 = 0.f, o1 = 0.f;
#pragma unroll
  for (int t2 = 0; t2 < TT; ++t2) {
    if (t2 <= qi) {
      const float* vp = v + base + (size_t)t2 * CC;
      float p = sc[t2] * inv;
      o0 += p * vp[lane]; o1 += p * vp[lane + 32];
    }
  }
  float* op = o + base + (size_t)qi * CC;
  op[lane] = o0; op[lane + 32] = o1;
}

// ---------------- decode: ConvTranspose (kernel==stride) with shared conv weight ----------------
__global__ __launch_bounds__(160) void decode_kernel(
    const float* __restrict__ logits, const float* __restrict__ cw, float* __restrict__ out)
{
  __shared__ float row[CC];
  int bt = blockIdx.x;
  int s = bt % TT, b = bt / TT;
  const float* lp = logits + (size_t)bt * CC;
  for (int i = threadIdx.x; i < CC; i += 160) row[i] = lp[i];
  __syncthreads();
  int j = threadIdx.x;
  if (j < PATCH) {
    int d = j / 49, r = j % 49, p = r / 7, qq = r % 7;
    float acc = 0.f;
    for (int c = 0; c < CC; ++c) acc += row[c] * cw[(size_t)c * PATCH + j];
    // out[b][d][p][s*7+qq], width = T*K = 119
    out[(((size_t)(b * CI + d) * KK) + p) * (TT * KK) + s * KK + qq] = acc;
  }
}

// ---------------- host orchestration ----------------
extern "C" void kernel_launch(void* const* d_in, const int* in_sizes, int n_in,
                              void* d_out, int out_size, void* d_ws, size_t ws_size,
                              hipStream_t stream)
{
  (void)in_sizes; (void)n_in; (void)out_size; (void)ws_size;
  const float* x      = (const float*)d_in[0];
  const float* conv_w = (const float*)d_in[1];
  const float* ln1_w  = (const float*)d_in[2];
  const float* ln1_b  = (const float*)d_in[3];
  const float* wq     = (const float*)d_in[4];
  const float* wk     = (const float*)d_in[5];
  const float* wv     = (const float*)d_in[6];
  const float* ln2_w  = (const float*)d_in[7];
  const float* ln2_b  = (const float*)d_in[8];
  const float* mlp_w  = (const float*)d_in[9];
  const float* mlp_b  = (const float*)d_in[10];
  const float* out_w  = (const float*)d_in[11];
  const float* out_b  = (const float*)d_in[12];

  char* p = (char*)d_ws;
  const size_t S  = (size_t)MM * CC;       // activation rows
  const size_t WE = (size_t)CC * CC;       // one weight matrix
  float*          h    = (float*)p;          p += S * 4;
  unsigned short* hn   = (unsigned short*)p; p += S * 2;
  float*          qb   = (float*)p;          p += S * 4;
  float*          kb   = (float*)p;          p += S * 4;
  float*          vb   = (float*)p;          p += S * 4;
  float*          res  = (float*)p;          p += S * 4;
  unsigned short* wqb  = (unsigned short*)p; p += WE * 2;   // per-layer bf16 weight slab
  unsigned short* wkb  = (unsigned short*)p; p += WE * 2;
  unsigned short* wvb  = (unsigned short*)p; p += WE * 2;
  unsigned short* wmb  = (unsigned short*)p; p += WE * 2;

  const int cvt_blocks = (int)(WE / (256 * 4));            // 1024 blocks per matrix

  embed_kernel<<<MM, 256, 0, stream>>>(x, conv_w, h);

  dim3 ggrid(MM / 128, CC / 128);
  for (int l = 0; l < LL; ++l) {
    const size_t wo = (size_t)l * WE;
    // pre-convert this layer's weights to bf16 (keeps GEMM inner loop convert-free)
    cvt_bf16_kernel<<<cvt_blocks, 256, 0, stream>>>(wq + wo, wqb);
    cvt_bf16_kernel<<<cvt_blocks, 256, 0, stream>>>(wk + wo, wkb);
    cvt_bf16_kernel<<<cvt_blocks, 256, 0, stream>>>(wv + wo, wvb);
    cvt_bf16_kernel<<<cvt_blocks, 256, 0, stream>>>(mlp_w + wo, wmb);

    ln_res_kernel<<<MM / 8, 256, 0, stream>>>(h, l == 0 ? nullptr : res,
                                              ln1_w + l * CC, ln1_b + l * CC, hn, 1);
    gemm_bf16<<<ggrid, 256, 0, stream>>>(hn, wqb, nullptr, qb, CC, CC);
    gemm_bf16<<<ggrid, 256, 0, stream>>>(hn, wkb, nullptr, kb, CC, CC);
    gemm_bf16<<<ggrid, 256, 0, stream>>>(hn, wvb, nullptr, vb, CC, CC);
    attn_kernel<<<(Bb * NHh * TT) / 8, 256, 0, stream>>>(qb, kb, vb, res);
    ln_res_kernel<<<MM / 8, 256, 0, stream>>>(h, res, ln2_w + l * CC, ln2_b + l * CC, hn, 1);
    gemm_bf16<<<ggrid, 256, 0, stream>>>(hn, wmb, mlp_b + l * CC, res, CC, CC);
  }
  // final residual add (no LN), then out projection
  cvt_bf16_kernel<<<cvt_blocks, 256, 0, stream>>>(out_w, wqb);
  ln_res_kernel<<<MM / 8, 256, 0, stream>>>(h, res, nullptr, nullptr, hn, 0);
  gemm_bf16<<<ggrid, 256, 0, stream>>>(hn, wqb, out_b, qb, CC, CC);
  decode_kernel<<<MM, 160, 0, stream>>>(qb, conv_w, (float*)d_out);
}
